// AttentionBlock_6734508720607
// MI455X (gfx1250) — compile-verified, matrix operand order checked by reading, not run on hardware
//
#include <hip/hip_runtime.h>
#include <hip/hip_bf16.h>
#include <math.h>

// ---------------------------------------------------------------------------
// AttentionBlock for MI455X (gfx1250): GroupNorm -> QKV -> MHSA -> proj + res
// B=4, C=1024, T=1024, 32 groups, 16 heads, head_dim=64.
// GEMMs via v_wmma_f32_16x16x32_f16; f16 tile staging via
// global_load_async_to_lds_b128 (ASYNCcnt) with ping-pong LDS buffers.
// ---------------------------------------------------------------------------

typedef _Float16 half_t;
typedef __attribute__((ext_vector_type(16))) _Float16 v16h;
typedef __attribute__((ext_vector_type(8)))  _Float16 v8h;
typedef __attribute__((ext_vector_type(8)))  float    v8f;

#define B_   4
#define C_   1024
#define T_   1024
#define NH_  16
#define CH_  64
#define GRP_ 32

// ---- CDNA5 async global->LDS copy (16B per lane), tracked by ASYNCcnt -----
__device__ __forceinline__ void async_b128(const half_t* gsrc, const _Float16* ldst) {
    const unsigned lds_addr = (unsigned)(uintptr_t)ldst;   // low 32b = LDS offset
    asm volatile("global_load_async_to_lds_b128 %0, %1, off"
                 :: "v"(lds_addr), "v"(gsrc) : "memory");
}
__device__ __forceinline__ void wait_async0() {
    asm volatile("s_wait_asynccnt 0x0" ::: "memory");
}

__device__ __forceinline__ v16h load_frag(const _Float16* row, int g) {
    // WMMA f16 A/B fragment: elems 0..7 hold k = 8g..8g+7,
    // elems 8..15 hold k = 16+8g..23+8g  (row points at [m][k0] base).
    const v8h lo = *reinterpret_cast<const v8h*>(row + (g << 3));
    const v8h hi = *reinterpret_cast<const v8h*>(row + 16 + (g << 3));
    v16h r;
#pragma unroll
    for (int i = 0; i < 8; ++i) { r[i] = lo[i]; r[i + 8] = hi[i]; }
    return r;
}

__device__ __forceinline__ v8f wmma16(v16h a, v16h b, v8f c) {
    return __builtin_amdgcn_wmma_f32_16x16x32_f16(false, a, false, b,
                                                  (short)0, c, false, false);
}

// ---------------------------------------------------------------------------
// Kernel 1: GroupNorm statistics.  grid = B*GRP_ blocks, 256 threads.
// ---------------------------------------------------------------------------
__global__ __launch_bounds__(256) void gn_stats_kernel(const float* __restrict__ x,
                                                       float* __restrict__ stats) {
    const int bg  = blockIdx.x;            // b*32 + group
    const int b   = bg >> 5;
    const int grp = bg & 31;
    const float* base = x + ((size_t)b * C_ + (size_t)grp * 32) * T_;

    float s = 0.f, ss = 0.f;
    for (int i = threadIdx.x; i < 32 * T_; i += 256) {
        float v = base[i];
        s += v; ss += v * v;
    }
#pragma unroll
    for (int m = 16; m >= 1; m >>= 1) {
        s  += __shfl_xor(s,  m, 32);
        ss += __shfl_xor(ss, m, 32);
    }
    __shared__ float ps[8][2];
    const int lane = threadIdx.x & 31, w = threadIdx.x >> 5;
    if (lane == 0) { ps[w][0] = s; ps[w][1] = ss; }
    __syncthreads();
    if (threadIdx.x == 0) {
        float ts = 0.f, tss = 0.f;
#pragma unroll
        for (int i = 0; i < 8; ++i) { ts += ps[i][0]; tss += ps[i][1]; }
        const float inv  = 1.0f / (32.0f * (float)T_);
        const float mean = ts * inv;
        const float var  = tss * inv - mean * mean;
        stats[bg * 2 + 0] = mean;
        stats[bg * 2 + 1] = rsqrtf(var + 1e-5f);
    }
}

// ---------------------------------------------------------------------------
// Kernel 2: apply GroupNorm affine, convert to f16, transpose to xn[b][t][c].
// grid = (T/64, C/64, B), 256 threads per 64x64 tile.
// ---------------------------------------------------------------------------
__global__ __launch_bounds__(256) void gn_apply_kernel(const float* __restrict__ x,
                                                       const float* __restrict__ gnw,
                                                       const float* __restrict__ gnb,
                                                       const float* __restrict__ stats,
                                                       half_t* __restrict__ xn) {
    __shared__ _Float16 ldsT[64][64];      // [t_local][c_local]
    const int t0 = blockIdx.x * 64;
    const int c0 = blockIdx.y * 64;
    const int b  = blockIdx.z;
    const int tid = threadIdx.x;

    const int c_local = tid >> 2;          // 0..63
    const int tseg    = (tid & 3) * 16;    // 0,16,32,48
    const int c       = c0 + c_local;
    const float mean = stats[(b * GRP_ + (c >> 5)) * 2 + 0];
    const float rstd = stats[(b * GRP_ + (c >> 5)) * 2 + 1];
    const float wgt  = gnw[c] * rstd;
    const float bia  = gnb[c] - mean * wgt;

    const float* src = x + ((size_t)b * C_ + c) * T_ + t0 + tseg;
#pragma unroll
    for (int i = 0; i < 16; i += 4) {
        float4 f = *reinterpret_cast<const float4*>(src + i);
        ldsT[tseg + i + 0][c_local] = (_Float16)(f.x * wgt + bia);
        ldsT[tseg + i + 1][c_local] = (_Float16)(f.y * wgt + bia);
        ldsT[tseg + i + 2][c_local] = (_Float16)(f.z * wgt + bia);
        ldsT[tseg + i + 3][c_local] = (_Float16)(f.w * wgt + bia);
    }
    __syncthreads();
    const int t_local = tid >> 2;
    const int cseg    = (tid & 3) * 16;
    half_t* dst = xn + ((size_t)b * T_ + t0 + t_local) * C_ + c0 + cseg;
    *reinterpret_cast<uint4*>(dst + 0) = *reinterpret_cast<const uint4*>(&ldsT[t_local][cseg + 0]);
    *reinterpret_cast<uint4*>(dst + 8) = *reinterpret_cast<const uint4*>(&ldsT[t_local][cseg + 8]);
}

// ---------------------------------------------------------------------------
// Kernel 3: QKV GEMM.  D[3072 x T] = qkv_w[3072 x 1024] * xn^T + bias,
// scattered into q/k layout [b*NH][t][64] (scaled by ch^-0.25) and
// v layout [b*NH][64][s].  grid = (24, 8, B), 256 threads (8 waves).
// Double-buffered K-loop; B tiles staged with async global->LDS.
// ---------------------------------------------------------------------------
__global__ __launch_bounds__(256, 2) void qkv_gemm_kernel(const float* __restrict__ W,
                                                          const float* __restrict__ bias,
                                                          const half_t* __restrict__ xn,
                                                          half_t* __restrict__ q,
                                                          half_t* __restrict__ k,
                                                          half_t* __restrict__ v) {
    __shared__ _Float16 lA[2][128][32];    // [m][k]  (weights, f32->f16)
    __shared__ _Float16 lB[2][128][32];    // [n][k]  (tokens)
    const int m0 = blockIdx.x * 128;
    const int n0 = blockIdx.y * 128;
    const int b  = blockIdx.z;
    const int tid  = threadIdx.x;
    const int lane = tid & 31, w = tid >> 5;
    const int wm = w & 3, wn = w >> 2;     // wave -> rows [wm*32,+32), cols [wn*64,+64)
    const int g = lane >> 4, ln = lane & 15;

    v8f acc[2][4];
#pragma unroll
    for (int i = 0; i < 2; ++i)
#pragma unroll
        for (int j = 0; j < 4; ++j) { v8f z = {}; acc[i][j] = z; }

    const int srow = tid >> 1;             // 0..127
    const int sseg = (tid & 1) * 16;       // 0 or 16

    auto stage_A = [&](int buf, int k0) {
        const float* src = W + (size_t)(m0 + srow) * C_ + k0 + sseg;
#pragma unroll
        for (int i = 0; i < 16; i += 4) {
            float4 f = *reinterpret_cast<const float4*>(src + i);
            lA[buf][srow][sseg + i + 0] = (_Float16)f.x;
            lA[buf][srow][sseg + i + 1] = (_Float16)f.y;
            lA[buf][srow][sseg + i + 2] = (_Float16)f.z;
            lA[buf][srow][sseg + i + 3] = (_Float16)f.w;
        }
    };
    auto stage_B = [&](int buf, int k0) {
        const half_t* bsrc = xn + ((size_t)b * T_ + n0 + srow) * C_ + k0 + sseg;
        async_b128(bsrc + 0, &lB[buf][srow][sseg + 0]);
        async_b128(bsrc + 8, &lB[buf][srow][sseg + 8]);
    };

    stage_A(0, 0);
    stage_B(0, 0);
    wait_async0();
    __syncthreads();

    for (int k0 = 0; k0 < C_; k0 += 32) {
        const int cur = (k0 >> 5) & 1;
        if (k0 + 32 < C_) {                // prefetch next tile into other buffer
            stage_A(cur ^ 1, k0 + 32);
            stage_B(cur ^ 1, k0 + 32);
        }
        v16h af[2], bf[4];
#pragma unroll
        for (int mt = 0; mt < 2; ++mt) af[mt] = load_frag(&lA[cur][wm * 32 + mt * 16 + ln][0], g);
#pragma unroll
        for (int nt = 0; nt < 4; ++nt) bf[nt] = load_frag(&lB[cur][wn * 64 + nt * 16 + ln][0], g);
#pragma unroll
        for (int mt = 0; mt < 2; ++mt)
#pragma unroll
            for (int nt = 0; nt < 4; ++nt)
                acc[mt][nt] = wmma16(af[mt], bf[nt], acc[mt][nt]);
        wait_async0();
        __syncthreads();
    }

    const float qk_scale = 0.35355339059327373f;  // 64^(-1/4)
#pragma unroll
    for (int mt = 0; mt < 2; ++mt)
#pragma unroll
        for (int nt = 0; nt < 4; ++nt)
#pragma unroll
            for (int e = 0; e < 8; ++e) {
                const int o     = m0 + wm * 32 + mt * 16 + e + 8 * g;
                const int token = n0 + wn * 64 + nt * 16 + ln;
                const float val = acc[mt][nt][e] + bias[o];
                const int h = o / 192, r = o % 192;
                const size_t bh = (size_t)b * NH_ + h;
                if (r < 64)
                    q[(bh * T_ + token) * CH_ + r] = (half_t)(val * qk_scale);
                else if (r < 128)
                    k[(bh * T_ + token) * CH_ + (r - 64)] = (half_t)(val * qk_scale);
                else
                    v[(bh * CH_ + (r - 128)) * T_ + token] = (half_t)val;
            }
}

// ---------------------------------------------------------------------------
// Kernel 4: streaming (flash-style) attention.
// grid = (B*NH, T/64), 128 threads (4 waves); each wave owns 16 query rows.
// K/V tiles double-buffered via async global->LDS copies.
// ---------------------------------------------------------------------------
__global__ __launch_bounds__(128, 2) void attn_kernel(const half_t* __restrict__ q,
                                                      const half_t* __restrict__ k,
                                                      const half_t* __restrict__ v,
                                                      half_t* __restrict__ a) {
    __shared__ _Float16 lQ[64][64];        // [t][ch]   A-layout for S
    __shared__ _Float16 lK[2][64][64];     // [s][ch]   B(T)-layout for S
    __shared__ _Float16 lV[2][64][64];     // [ch][s]   B(T)-layout for P*V
    __shared__ _Float16 lP[4][16][64];     // per-wave P strip [m][s]

    const int bh = blockIdx.x;
    const int t0 = blockIdx.y * 64;
    const int tid  = threadIdx.x;
    const int lane = tid & 31, w = tid >> 5;
    const int g = lane >> 4, ln = lane & 15;

    const half_t* qb = q + (size_t)bh * T_ * CH_;
    const half_t* kb = k + (size_t)bh * T_ * CH_;
    const half_t* vb = v + (size_t)bh * CH_ * T_;

    const int srow = tid >> 1;             // 0..63
    const int sseg = (tid & 1) * 32;       // 0 or 32

    auto stage_KV = [&](int buf, int s0) {
        const half_t* ksrc = kb + (size_t)(s0 + srow) * CH_ + sseg;
        const half_t* vsrc = vb + (size_t)srow * T_ + s0 + sseg;
#pragma unroll
        for (int i = 0; i < 4; ++i) {
            async_b128(ksrc + i * 8, &lK[buf][srow][sseg + i * 8]);
            async_b128(vsrc + i * 8, &lV[buf][srow][sseg + i * 8]);
        }
    };

    {   // stage Q tile once (async) + first K/V tile
        const half_t* src = qb + (size_t)(t0 + srow) * CH_ + sseg;
#pragma unroll
        for (int i = 0; i < 4; ++i)
            async_b128(src + i * 8, &lQ[srow][sseg + i * 8]);
        stage_KV(0, 0);
    }
    wait_async0();
    __syncthreads();

    v16h qa[2];
#pragma unroll
    for (int ks = 0; ks < 2; ++ks) qa[ks] = load_frag(&lQ[w * 16 + ln][ks * 32], g);

    float rmax[8], rsum[8];
    v8f   o[4];
#pragma unroll
    for (int e = 0; e < 8; ++e) { rmax[e] = -1e30f; rsum[e] = 0.f; }
#pragma unroll
    for (int nt = 0; nt < 4; ++nt) { v8f z = {}; o[nt] = z; }

    for (int s0 = 0; s0 < T_; s0 += 64) {
        const int cur = (s0 >> 6) & 1;
        if (s0 + 64 < T_) stage_KV(cur ^ 1, s0 + 64);   // prefetch next K/V tile

        // S strip: 16 queries x 64 keys
        v8f sf[4];
#pragma unroll
        for (int nt = 0; nt < 4; ++nt) { v8f z = {}; sf[nt] = z; }
#pragma unroll
        for (int ks = 0; ks < 2; ++ks)
#pragma unroll
            for (int nt = 0; nt < 4; ++nt) {
                v16h bf = load_frag(&lK[cur][nt * 16 + ln][ks * 32], g);
                sf[nt] = wmma16(qa[ks], bf, sf[nt]);
            }

        // online softmax: row m = e + 8g, row lives in one 16-lane group
#pragma unroll
        for (int e = 0; e < 8; ++e) {
            float tmax = fmaxf(fmaxf(sf[0][e], sf[1][e]), fmaxf(sf[2][e], sf[3][e]));
#pragma unroll
            for (int m = 1; m <= 8; m <<= 1) tmax = fmaxf(tmax, __shfl_xor(tmax, m, 32));
            const float nm = fmaxf(rmax[e], tmax);
            const float sc = __expf(rmax[e] - nm);
            rmax[e] = nm;
            rsum[e] *= sc;
#pragma unroll
            for (int nt = 0; nt < 4; ++nt) o[nt][e] *= sc;
            float ps = 0.f;
#pragma unroll
            for (int nt = 0; nt < 4; ++nt) {
                const float p = __expf(sf[nt][e] - nm);
                sf[nt][e] = p;
                ps += p;
            }
#pragma unroll
            for (int m = 1; m <= 8; m <<= 1) ps += __shfl_xor(ps, m, 32);
            rsum[e] += ps;
        }

        // spill P (D layout) to wave-private LDS strip, reload as A fragments
#pragma unroll
        for (int nt = 0; nt < 4; ++nt)
#pragma unroll
            for (int e = 0; e < 8; ++e)
                lP[w][e + 8 * g][nt * 16 + ln] = (_Float16)sf[nt][e];
        __syncthreads();

#pragma unroll
        for (int ks = 0; ks < 2; ++ks) {
            v16h pa = load_frag(&lP[w][ln][ks * 32], g);
#pragma unroll
            for (int nt = 0; nt < 4; ++nt) {
                v16h bf = load_frag(&lV[cur][nt * 16 + ln][ks * 32], g);
                o[nt] = wmma16(pa, bf, o[nt]);
            }
        }
        wait_async0();     // next tile fully in LDS
        __syncthreads();   // all waves done reading current buffers
    }

    // write a[b][t][c] (token-major so proj GEMM stages contiguously)
    const int b = bh >> 4, h = bh & 15;
#pragma unroll
    for (int nt = 0; nt < 4; ++nt)
#pragma unroll
        for (int e = 0; e < 8; ++e) {
            const int t  = t0 + w * 16 + e + 8 * g;
            const int cd = nt * 16 + ln;
            a[((size_t)b * T_ + t) * C_ + h * CH_ + cd] =
                (half_t)(o[nt][e] / rsum[e]);
        }
}

// ---------------------------------------------------------------------------
// Kernel 5: output projection + bias + residual.  grid = (8, 8, B), 256 thr.
// ---------------------------------------------------------------------------
__global__ __launch_bounds__(256, 2) void proj_gemm_kernel(const float* __restrict__ W,
                                                           const float* __restrict__ bias,
                                                           const half_t* __restrict__ a,
                                                           const float* __restrict__ x,
                                                           float* __restrict__ out) {
    __shared__ _Float16 lA[2][128][32];
    __shared__ _Float16 lB[2][128][32];
    const int m0 = blockIdx.x * 128;
    const int n0 = blockIdx.y * 128;
    const int b  = blockIdx.z;
    const int tid  = threadIdx.x;
    const int lane = tid & 31, w = tid >> 5;
    const int wm = w & 3, wn = w >> 2;
    const int g = lane >> 4, ln = lane & 15;

    v8f acc[2][4];
#pragma unroll
    for (int i = 0; i < 2; ++i)
#pragma unroll
        for (int j = 0; j < 4; ++j) { v8f z = {}; acc[i][j] = z; }

    const int srow = tid >> 1;
    const int sseg = (tid & 1) * 16;

    auto stage_A = [&](int buf, int k0) {
        const float* src = W + (size_t)(m0 + srow) * C_ + k0 + sseg;
#pragma unroll
        for (int i = 0; i < 16; i += 4) {
            float4 f = *reinterpret_cast<const float4*>(src + i);
            lA[buf][srow][sseg + i + 0] = (_Float16)f.x;
            lA[buf][srow][sseg + i + 1] = (_Float16)f.y;
            lA[buf][srow][sseg + i + 2] = (_Float16)f.z;
            lA[buf][srow][sseg + i + 3] = (_Float16)f.w;
        }
    };
    auto stage_B = [&](int buf, int k0) {
        const half_t* bsrc = a + ((size_t)b * T_ + n0 + srow) * C_ + k0 + sseg;
        async_b128(bsrc + 0, &lB[buf][srow][sseg + 0]);
        async_b128(bsrc + 8, &lB[buf][srow][sseg + 8]);
    };

    stage_A(0, 0);
    stage_B(0, 0);
    wait_async0();
    __syncthreads();

    for (int k0 = 0; k0 < C_; k0 += 32) {
        const int cur = (k0 >> 5) & 1;
        if (k0 + 32 < C_) {
            stage_A(cur ^ 1, k0 + 32);
            stage_B(cur ^ 1, k0 + 32);
        }
        v16h af[2], bf[4];
#pragma unroll
        for (int mt = 0; mt < 2; ++mt) af[mt] = load_frag(&lA[cur][wm * 32 + mt * 16 + ln][0], g);
#pragma unroll
        for (int nt = 0; nt < 4; ++nt) bf[nt] = load_frag(&lB[cur][wn * 64 + nt * 16 + ln][0], g);
#pragma unroll
        for (int mt = 0; mt < 2; ++mt)
#pragma unroll
            for (int nt = 0; nt < 4; ++nt)
                acc[mt][nt] = wmma16(af[mt], bf[nt], acc[mt][nt]);
        wait_async0();
        __syncthreads();
    }

#pragma unroll
    for (int mt = 0; mt < 2; ++mt)
#pragma unroll
        for (int nt = 0; nt < 4; ++nt)
#pragma unroll
            for (int e = 0; e < 8; ++e) {
                const int oc    = m0 + wm * 32 + mt * 16 + e + 8 * g;
                const int token = n0 + wn * 64 + nt * 16 + ln;
                const size_t idx = ((size_t)b * C_ + oc) * T_ + token;
                out[idx] = acc[mt][nt][e] + bias[oc] + x[idx];
            }
}

// ---------------------------------------------------------------------------
// Host-side launcher
// ---------------------------------------------------------------------------
extern "C" void kernel_launch(void* const* d_in, const int* in_sizes, int n_in,
                              void* d_out, int out_size, void* d_ws, size_t ws_size,
                              hipStream_t stream) {
    const float* x      = (const float*)d_in[0];
    const float* gnw    = (const float*)d_in[1];
    const float* gnb    = (const float*)d_in[2];
    const float* qkv_w  = (const float*)d_in[3];
    const float* qkv_b  = (const float*)d_in[4];
    const float* proj_w = (const float*)d_in[5];
    const float* proj_b = (const float*)d_in[6];
    float* out = (float*)d_out;

    char* ws = (char*)d_ws;
    const size_t PLANE = (size_t)B_ * C_ * T_ * sizeof(half_t);  // 8 MiB
    float*  stats = (float*)ws;
    half_t* xn = (half_t*)(ws + 4096);
    half_t* q  = (half_t*)(ws + 4096 + 1 * PLANE);
    half_t* k  = (half_t*)(ws + 4096 + 2 * PLANE);
    half_t* v  = (half_t*)(ws + 4096 + 3 * PLANE);
    half_t* a  = (half_t*)(ws + 4096 + 4 * PLANE);
    (void)in_sizes; (void)n_in; (void)out_size; (void)ws_size;

    gn_stats_kernel<<<B_ * GRP_, 256, 0, stream>>>(x, stats);
    gn_apply_kernel<<<dim3(T_ / 64, C_ / 64, B_), 256, 0, stream>>>(x, gnw, gnb, stats, xn);
    qkv_gemm_kernel<<<dim3(3 * C_ / 128, T_ / 128, B_), 256, 0, stream>>>(qkv_w, qkv_b, xn, q, k, v);
    attn_kernel<<<dim3(B_ * NH_, T_ / 64), 128, 0, stream>>>(q, k, v, a);
    proj_gemm_kernel<<<dim3(C_ / 128, T_ / 128, B_), 256, 0, stream>>>(proj_w, proj_b, a, x, out);
}
